// CausalSelfAttention_3770981286022
// MI455X (gfx1250) — matrix operand order repaired and checked by reading, not compiled, not run on hardware
//
#include <hip/hip_runtime.h>
#include <hip/hip_bf16.h>

// Causal self-attention, CDNA5 (gfx1250), wave32, WMMA bf16 path.
//   1) wconv:     W_attn, W_proj fp32 -> bf16, transposed to [N][K]
//   2) qkv_gemm:  x @ W_attn + b -> Q[B,H,T,D], K[B,H,T,D], V^T[B,H,D,T] (bf16)
//   3) attn:      flash-attention (online softmax, fp32 accum) -> y [B,T,C] bf16
//   4) proj_gemm: y @ W_proj + b -> out fp32
//
// WMMA operand layouts (wave32, v_wmma_f32_16x16x32_bf16):
//   A (16x32): lane l -> row M = l%16; k = 8*(l/16)+{0..7} (regs 0-3),
//              16+8*(l/16)+{0..7} (regs 4-7)  => two 16B contiguous loads/lane
//   B (32x16): lane l -> col N = l%16; k = 8*(l/16)+{0..7}, 16+8*(l/16)+{0..7} (as A) => two 16B loads/lane
//   C/D f32  : lane l, vreg j -> (M = j + 8*(l/16), N = l%16)
//
// Tiling: 32-row M tiles per wave so every B operand feeds two WMMAs
// (GEMMs: 8 WMMA / 8 B-loads per 32-K step; attention: 16 WMMA per 32-key
// step with K/V operands loaded once -> halves L2 K/V traffic).

#define BB 4
#define TT 2048
#define CC 1024
#define HH 16
#define DD 64

typedef __attribute__((ext_vector_type(16))) __bf16 v16bf;
typedef __attribute__((ext_vector_type(8)))  __bf16 bf16x8;
typedef __attribute__((ext_vector_type(8)))  float  v8f;

union Op16 { v16bf v; bf16x8 h2[2]; __bf16 e[16]; };

__device__ __forceinline__ v8f wmma_bf16(v16bf a, v16bf b, v8f c) {
  return __builtin_amdgcn_wmma_f32_16x16x32_bf16(
      /*neg_a=*/false, a, /*neg_b=*/false, b,
      /*c_mod=*/(short)0, c, /*reuse_a=*/false, /*reuse_b=*/false);
}

__device__ __forceinline__ bf16x8 ld8(const __bf16* p) {
  return *reinterpret_cast<const bf16x8*>(p);
}

// A operand from row-major bf16 (contraction contiguous). p = row + kbase.
__device__ __forceinline__ v16bf a_from_bf16(const __bf16* p, int h) {
  Op16 u;
  u.h2[0] = ld8(p + 8 * h);        // k = 8h .. 8h+7
  u.h2[1] = ld8(p + 16 + 8 * h);   // k = 16+8h .. 16+8h+7
  return u.v;
}

// A operand from row-major fp32 with on-the-fly bf16 convert. p = row + kbase.
__device__ __forceinline__ v16bf a_from_f32(const float* p, int h) {
  Op16 u;
  const float4* q0 = reinterpret_cast<const float4*>(p + 8 * h);
  const float4* q1 = reinterpret_cast<const float4*>(p + 16 + 8 * h);
  float4 f0 = q0[0], f1 = q0[1], f2 = q1[0], f3 = q1[1];
  u.e[0]=(__bf16)f0.x; u.e[1]=(__bf16)f0.y; u.e[2]=(__bf16)f0.z; u.e[3]=(__bf16)f0.w;
  u.e[4]=(__bf16)f1.x; u.e[5]=(__bf16)f1.y; u.e[6]=(__bf16)f1.z; u.e[7]=(__bf16)f1.w;
  u.e[8]=(__bf16)f2.x; u.e[9]=(__bf16)f2.y; u.e[10]=(__bf16)f2.z; u.e[11]=(__bf16)f2.w;
  u.e[12]=(__bf16)f3.x; u.e[13]=(__bf16)f3.y; u.e[14]=(__bf16)f3.z; u.e[15]=(__bf16)f3.w;
  return u.v;
}

// B operand: this lane's column is a contiguous row in memory (transposed
// weights / K rows / V^T rows). p = column-row base + kbase.
__device__ __forceinline__ v16bf b_from_rows(const __bf16* p, int h) {
  Op16 u;
  u.h2[0] = ld8(p + 8 * h);        // k = 8h .. 8h+7
  u.h2[1] = ld8(p + 16 + 8 * h);   // k = 16+8h .. 16+8h+7
  return u.v;
}

__device__ __forceinline__ float rmax16(float v) {
  v = fmaxf(v, __shfl_xor(v, 1, 16));
  v = fmaxf(v, __shfl_xor(v, 2, 16));
  v = fmaxf(v, __shfl_xor(v, 4, 16));
  v = fmaxf(v, __shfl_xor(v, 8, 16));
  return v;
}
__device__ __forceinline__ float rsum16(float v) {
  v += __shfl_xor(v, 1, 16);
  v += __shfl_xor(v, 2, 16);
  v += __shfl_xor(v, 4, 16);
  v += __shfl_xor(v, 8, 16);
  return v;
}

// ---------------- 1) weight convert + transpose (fp32 [K][N] -> bf16 [N][K])
__global__ void csa_wconv(const float* __restrict__ W, __bf16* __restrict__ Wt,
                          int K, int N) {
  int idx = blockIdx.x * blockDim.x + threadIdx.x;
  if (idx < K * N) {
    int k = idx / N, n = idx % N;                 // coalesced read over n
    Wt[(size_t)n * K + k] = (__bf16)W[idx];
  }
}

// ---------------- 2) QKV projection GEMM (M=8192, N=3072, K=1024)
// block = 256 (8 waves); wave tile = 32 rows x 64 cols; grid = (6, 256)
__global__ void csa_qkv(const float* __restrict__ x,
                        const __bf16* __restrict__ Wt,     // [3C][C]
                        const float* __restrict__ bias,    // [3C]
                        __bf16* __restrict__ Qb,           // [B,H,T,D]
                        __bf16* __restrict__ Kb,           // [B,H,T,D]
                        __bf16* __restrict__ Vt) {         // [B,H,D,T]
  const int w  = __builtin_amdgcn_readfirstlane(threadIdx.x >> 5);
  const int lane = threadIdx.x & 31;
  const int lh = lane >> 4, ln = lane & 15;
  const int m0 = blockIdx.y * 32;
  const int n0 = (blockIdx.x * 8 + w) * 64;

  v8f zero = {};
  v8f acc[2][4] = {{zero, zero, zero, zero}, {zero, zero, zero, zero}};

  const float* arow0 = x + (size_t)(m0 + ln) * CC;
  const float* arow1 = x + (size_t)(m0 + 16 + ln) * CC;
  for (int kk = 0; kk < CC; kk += 32) {
    v16bf a0 = a_from_f32(arow0 + kk, lh);
    v16bf a1 = a_from_f32(arow1 + kk, lh);
#pragma unroll
    for (int nt = 0; nt < 4; ++nt) {
      const __bf16* brow = Wt + (size_t)(n0 + nt * 16 + ln) * CC + kk;
      v16bf b = b_from_rows(brow, lh);
      acc[0][nt] = wmma_bf16(a0, b, acc[0][nt]);
      acc[1][nt] = wmma_bf16(a1, b, acc[1][nt]);
    }
  }

#pragma unroll
  for (int nt = 0; nt < 4; ++nt) {
    // tile base column is wave-uniform; scalarize so region split is SALU
    const int cbase  = __builtin_amdgcn_readfirstlane(n0 + nt * 16);
    const int region = cbase >> 10;                // 0=Q 1=K 2=V
    const int hh     = (cbase >> 6) & (HH - 1);
    const int dd     = (cbase & 63) + ln;
    const float bv   = bias[cbase + ln];
#pragma unroll
    for (int mt = 0; mt < 2; ++mt)
#pragma unroll
      for (int j = 0; j < 8; ++j) {
        const int row = m0 + mt * 16 + j + 8 * lh; // global token index
        const int bb = row >> 11, tt = row & (TT - 1);
        const __bf16 val = (__bf16)(acc[mt][nt][j] + bv);
        const size_t bh = (size_t)(bb * HH + hh);
        if (region == 0)      Qb[(bh * TT + tt) * DD + dd] = val;
        else if (region == 1) Kb[(bh * TT + tt) * DD + dd] = val;
        else                  Vt[(bh * DD + dd) * TT + tt] = val;
      }
  }
}

// ---------------- 3) flash attention; grid = (B*H, T/128), block = 128 (4 waves)
// wave tile = 32 queries; K/V B-operands loaded once per 32-key step, shared
// by both 16-row query subtiles (16 WMMAs per step).
__global__ void csa_attn(const __bf16* __restrict__ Qb,
                         const __bf16* __restrict__ Kb,
                         const __bf16* __restrict__ Vt,
                         __bf16* __restrict__ yb) {        // [B,T,C]
  const int bh = blockIdx.x;
  const int b = bh >> 4, hd = bh & 15;
  const int w = __builtin_amdgcn_readfirstlane(threadIdx.x >> 5);
  const int lane = threadIdx.x & 31;
  const int lh = lane >> 4, ln = lane & 15;
  const int q0 = blockIdx.y * 128 + w * 32;

  const __bf16* Qbase = Qb + (size_t)bh * TT * DD;
  const __bf16* Kbase = Kb + (size_t)bh * TT * DD;
  const __bf16* Vbase = Vt + (size_t)bh * DD * TT;

  // Q A-operands: 2 row-subtiles x 2 d-slices, reused across whole key loop
  v16bf aq[2][2];
#pragma unroll
  for (int mt = 0; mt < 2; ++mt) {
    const __bf16* qrow = Qbase + (size_t)(q0 + mt * 16 + ln) * DD;
    aq[mt][0] = a_from_bf16(qrow, lh);        // d = 0..31
    aq[mt][1] = a_from_bf16(qrow + 32, lh);   // d = 32..63
  }

  float m[2][8], lsum[2][8], alpha[2][8];
  v8f zero = {};
  v8f o[2][4] = {{zero, zero, zero, zero}, {zero, zero, zero, zero}};
#pragma unroll
  for (int mt = 0; mt < 2; ++mt)
#pragma unroll
    for (int j = 0; j < 8; ++j) { m[mt][j] = -3.0e38f; lsum[mt][j] = 0.0f; }

  __shared__ __align__(16) __bf16 plds[4][2][16][32];  // per-wave P staging

  const float scale = 0.125f;                          // 1/sqrt(64)
  for (int kt = 0; kt <= q0 + 31; kt += 32) {
    // --- S = Q K^T for 32 keys (two 16-col subtiles), D=64
    v16bf bk[2][2];
#pragma unroll
    for (int ss = 0; ss < 2; ++ss) {
      const __bf16* krow = Kbase + (size_t)(kt + ss * 16 + ln) * DD;
      bk[ss][0] = b_from_rows(krow, lh);
      bk[ss][1] = b_from_rows(krow + 32, lh);
    }
    v8f s[2][2];
#pragma unroll
    for (int mt = 0; mt < 2; ++mt)
#pragma unroll
      for (int ss = 0; ss < 2; ++ss)
        s[mt][ss] = wmma_bf16(aq[mt][1], bk[ss][1],
                              wmma_bf16(aq[mt][0], bk[ss][0], zero));

    // --- scale + causal mask + online softmax update
#pragma unroll
    for (int mt = 0; mt < 2; ++mt)
#pragma unroll
      for (int j = 0; j < 8; ++j) {
        const int q  = q0 + mt * 16 + j + 8 * lh;
        const int k0 = kt + ln, k1 = kt + 16 + ln;
        float v0 = (k0 <= q) ? s[mt][0][j] * scale : -3.0e38f;
        float v1 = (k1 <= q) ? s[mt][1][j] * scale : -3.0e38f;
        const float tm = rmax16(fmaxf(v0, v1));
        const float mn = fmaxf(m[mt][j], tm);
        alpha[mt][j] = __expf(m[mt][j] - mn);
        m[mt][j] = mn;
        const float p0 = __expf(v0 - mn);
        const float p1 = __expf(v1 - mn);
        lsum[mt][j] = lsum[mt][j] * alpha[mt][j] + rsum16(p0 + p1);
        s[mt][0][j] = p0; s[mt][1][j] = p1;
      }
#pragma unroll
    for (int mt = 0; mt < 2; ++mt)
#pragma unroll
      for (int dt = 0; dt < 4; ++dt)
#pragma unroll
        for (int j = 0; j < 8; ++j) o[mt][dt][j] *= alpha[mt][j];

    // --- P (C-layout fp32) -> bf16 -> LDS -> reload in A-operand layout.
    // LDS region is wave-private; per-wave LDS ops are in-order (DScnt);
    // s_wait_dscnt 0 as belt-and-braces before the cross-lane readback.
#pragma unroll
    for (int mt = 0; mt < 2; ++mt)
#pragma unroll
      for (int ss = 0; ss < 2; ++ss)
#pragma unroll
        for (int j = 0; j < 8; ++j)
          plds[w][mt][j + 8 * lh][ss * 16 + ln] = (__bf16)s[mt][ss][j];
    __asm__ volatile("s_wait_dscnt 0" ::: "memory");
    v16bf ap[2];
    ap[0] = a_from_bf16(&plds[w][0][ln][0], lh);
    ap[1] = a_from_bf16(&plds[w][1][ln][0], lh);

    // --- O += P @ V (V^T rows contiguous over keys; one load, two WMMAs)
#pragma unroll
    for (int dt = 0; dt < 4; ++dt) {
      const __bf16* vrow = Vbase + (size_t)(dt * 16 + ln) * TT + kt;
      v16bf bv = b_from_rows(vrow, lh);
      o[0][dt] = wmma_bf16(ap[0], bv, o[0][dt]);
      o[1][dt] = wmma_bf16(ap[1], bv, o[1][dt]);
    }
  }

  // --- epilogue: O /= l, store y[B,T,C] bf16
#pragma unroll
  for (int mt = 0; mt < 2; ++mt)
#pragma unroll
    for (int j = 0; j < 8; ++j) {
      const float inv = 1.0f / lsum[mt][j];
      const int t = q0 + mt * 16 + j + 8 * lh;
      __bf16* yrow = yb + ((size_t)(b * TT + t)) * CC + hd * DD;
#pragma unroll
      for (int dt = 0; dt < 4; ++dt)
        yrow[dt * 16 + ln] = (__bf16)(o[mt][dt][j] * inv);
    }
}

// ---------------- 4) output projection GEMM (M=8192, N=1024, K=1024) -> fp32
// block = 256 (8 waves); wave tile = 32 rows x 64 cols; grid = (2, 256)
__global__ void csa_proj(const __bf16* __restrict__ yb,    // [B*T][C]
                         const __bf16* __restrict__ Wpt,   // [C][C] transposed
                         const float* __restrict__ bias,   // [C]
                         float* __restrict__ out) {        // [B*T][C] fp32
  const int w = __builtin_amdgcn_readfirstlane(threadIdx.x >> 5);
  const int lane = threadIdx.x & 31;
  const int lh = lane >> 4, ln = lane & 15;
  const int m0 = blockIdx.y * 32;
  const int n0 = (blockIdx.x * 8 + w) * 64;

  v8f zero = {};
  v8f acc[2][4] = {{zero, zero, zero, zero}, {zero, zero, zero, zero}};

  const __bf16* arow0 = yb + (size_t)(m0 + ln) * CC;
  const __bf16* arow1 = yb + (size_t)(m0 + 16 + ln) * CC;
  for (int kk = 0; kk < CC; kk += 32) {
    v16bf a0 = a_from_bf16(arow0 + kk, lh);
    v16bf a1 = a_from_bf16(arow1 + kk, lh);
#pragma unroll
    for (int nt = 0; nt < 4; ++nt) {
      const __bf16* brow = Wpt + (size_t)(n0 + nt * 16 + ln) * CC + kk;
      v16bf b = b_from_rows(brow, lh);
      acc[0][nt] = wmma_bf16(a0, b, acc[0][nt]);
      acc[1][nt] = wmma_bf16(a1, b, acc[1][nt]);
    }
  }

#pragma unroll
  for (int nt = 0; nt < 4; ++nt) {
    const int c = n0 + nt * 16 + ln;
    const float bv = bias[c];
#pragma unroll
    for (int mt = 0; mt < 2; ++mt)
#pragma unroll
      for (int j = 0; j < 8; ++j) {
        const int row = m0 + mt * 16 + j + 8 * lh;
        out[(size_t)row * CC + c] = acc[mt][nt][j] + bv;
      }
  }
}

extern "C" void kernel_launch(void* const* d_in, const int* in_sizes, int n_in,
                              void* d_out, int out_size, void* d_ws, size_t ws_size,
                              hipStream_t stream) {
  const float* x      = (const float*)d_in[0];   // [4,2048,1024]
  const float* W_attn = (const float*)d_in[1];   // [1024,3072]
  const float* b_attn = (const float*)d_in[2];   // [3072]
  const float* W_proj = (const float*)d_in[3];   // [1024,1024]
  const float* b_proj = (const float*)d_in[4];   // [1024]
  float* out = (float*)d_out;

  // workspace layout (bytes), total ~72 MB
  char* ws = (char*)d_ws;
  __bf16* Wqkvt = (__bf16*)(ws);                         //  6,291,456
  __bf16* Wpt   = (__bf16*)(ws + 6291456);               //  2,097,152
  __bf16* Qb    = (__bf16*)(ws + 8388608);               // 16,777,216
  __bf16* Kb    = (__bf16*)(ws + 25165824);              // 16,777,216
  __bf16* Vt    = (__bf16*)(ws + 41943040);              // 16,777,216
  __bf16* yb    = (__bf16*)(ws + 58720256);              // 16,777,216

  csa_wconv<<<(CC * 3 * CC + 255) / 256, 256, 0, stream>>>(W_attn, Wqkvt, CC, 3 * CC);
  csa_wconv<<<(CC * CC + 255) / 256, 256, 0, stream>>>(W_proj, Wpt, CC, CC);
  csa_qkv<<<dim3(6, 256), 256, 0, stream>>>(x, Wqkvt, b_attn, Qb, Kb, Vt);
  csa_attn<<<dim3(BB * HH, TT / 128), 128, 0, stream>>>(Qb, Kb, Vt, yb);
  csa_proj<<<dim3(2, 256), 256, 0, stream>>>(yb, Wpt, b_proj, out);
}
